// MSRResNetKPN_34626026340608
// MI455X (gfx1250) — compile-verified, matrix operand order checked by reading, not run on hardware
//
#include <hip/hip_runtime.h>
#include <hip/hip_bf16.h>

typedef __attribute__((ext_vector_type(16))) _Float16 v16h;
typedef __attribute__((ext_vector_type(8)))  float    v8f;

#define NPOS 262144        // B*H*W = 4*256*256
#define HH 256
#define WW 256
#define PW 258             // padded row width
#define PPLANE (258*258)   // 66564 positions per padded image plane
#define PADPOS (4*PPLANE)  // 266256

__device__ __forceinline__ int pad_idx(int n, int y, int x) {
  // interior position (y,x) of image n inside the zero-padded plane
  return n * PPLANE + (y + 1) * PW + (x + 1);
}

// ---------------------------------------------------------------------------
// Weight packing: OIHW fp32 -> WMMA B-fragment layout, f16, cout-tile innermost.
// flat index = ((tap*KC + kc)*CT + ct)*512 + lane*16 + j
//   col = lane&15 (cout in tile), k = (lane>>4)*16 + j, ci = kc*32 + k
// ---------------------------------------------------------------------------
__global__ __launch_bounds__(256)
void pack_w_kernel(const float* __restrict__ w, _Float16* __restrict__ wp,
                   int Cin, int Cout, int CT, int KC) {
  int idx = blockIdx.x * blockDim.x + threadIdx.x;
  int total = CT * 9 * KC * 512;
  if (idx >= total) return;
  int j    = idx & 15;
  int lane = (idx >> 4) & 31;
  int rest = idx >> 9;
  int ct   = rest % CT;
  int kc   = (rest / CT) % KC;
  int tap  = rest / (CT * KC);
  int co   = ct * 16 + (lane & 15);
  int ci   = kc * 32 + ((lane >> 4) << 4) + j;
  float v = 0.0f;
  if (co < Cout && ci < Cin)
    v = w[(size_t)(co * Cin + ci) * 9 + tap];
  wp[idx] = (_Float16)v;
}

// Zero the 1-pixel halo of a padded NHWC buffer (C channels).
__global__ __launch_bounds__(256)
void zero_halo_kernel(_Float16* __restrict__ buf, int C) {
  int i = blockIdx.x * blockDim.x + threadIdx.x;   // over PADPOS
  if (i >= PADPOS) return;
  int r = i % PPLANE;
  int y = r / PW, x = r % PW;
  if (y != 0 && y != PW - 1 && x != 0 && x != PW - 1) return;
  _Float16* p = buf + (size_t)i * C;
  for (int c = 0; c < C; ++c) p[c] = (_Float16)0.0f;
}

// ---------------------------------------------------------------------------
// Implicit-GEMM 3x3 SAME conv, WMMA f16 -> f32 accum, padded input.
// One wave computes a 16(M = x positions) x CT*16(N = cout) tile:
// per (tap,kc) a single A fragment feeds CT independent WMMAs.
// in : padded NHWC f16, C = KC*32. out: NHWC f16 (padded iff opad), stride ostride.
// res: optional residual (padded NHWC, stride rstride). relu: 0 none,1 relu,2 leaky.
// No divergent branches: EXEC is all-ones everywhere (grid is exact).
// ---------------------------------------------------------------------------
template <int KC, int CT>
__global__ __launch_bounds__(256)
void conv3x3p_kernel(const _Float16* __restrict__ in,
                     const _Float16* __restrict__ wp,
                     const float* __restrict__ bias, int bias_n,
                     _Float16* __restrict__ out, int ostride, int coff, int opad,
                     const _Float16* __restrict__ res, int rstride,
                     int relu) {
  const int CIN = KC * 32;
  int wave = blockIdx.x * (blockDim.x >> 5) + (threadIdx.x >> 5);  // 0..16383
  int l  = threadIdx.x & 31;
  int m  = l & 15;                 // x offset in tile / cout column
  int kh = l >> 4;                 // K-half

  int pos0 = wave << 4;
  int n = pos0 >> 16, y = (pos0 >> 8) & 255, x0 = pos0 & 255;

  v8f acc[CT];
#pragma unroll
  for (int ct = 0; ct < CT; ++ct) {
    int co = ct * 16 + m;
    float bv = (co < bias_n) ? bias[co] : 0.0f;
#pragma unroll
    for (int r = 0; r < 8; ++r) acc[ct][r] = bv;
  }

  const _Float16* wb = wp + (size_t)l * 16;

  for (int tap = 0; tap < 9; ++tap) {
    int dy = tap / 3 - 1, dx = tap % 3 - 1;
    // padded: always in-bounds, halo holds zeros
    const _Float16* ap =
        in + ((size_t)(n * PPLANE + (y + dy + 1) * PW + (x0 + dx + 1) + m)) * CIN
           + kh * 16;
    const _Float16* bp = wb + (size_t)tap * KC * CT * 512;
#pragma unroll
    for (int kc = 0; kc < KC; ++kc) {
      v16h a = *(const v16h*)(ap + kc * 32);             // one 32B load ...
#pragma unroll
      for (int ct = 0; ct < CT; ++ct) {                  // ... feeds CT WMMAs
        v16h b = *(const v16h*)(bp + (size_t)(kc * CT + ct) * 512);
        acc[ct] = __builtin_amdgcn_wmma_f32_16x16x32_f16(
            false, a, false, b, (short)0, acc[ct], false, false);
      }
    }
  }

  int pbase_pad = pad_idx(n, y, x0);                     // + pm is still interior
#pragma unroll
  for (int ct = 0; ct < CT; ++ct) {
    int co = ct * 16 + m;
#pragma unroll
    for (int r = 0; r < 8; ++r) {
      float v = acc[ct][r];
      if (relu == 1)      v = v > 0.0f ? v : 0.0f;
      else if (relu == 2) v = v > 0.0f ? v : 0.1f * v;
      int pm = r + (kh << 3);                            // D-layout M mapping
      if (res) v += (float)res[(size_t)(pbase_pad + pm) * rstride + co];
      size_t p = opad ? (size_t)(pbase_pad + pm) : (size_t)(pos0 + pm);
      out[p * ostride + coff + co] = (_Float16)v;
    }
  }
}

// ---------------------------------------------------------------------------
// Cin=3 conv (conv_first / c3_1): fp32 NCHW x -> padded f16 NHWC-64.
// mode 1=relu, 2=leaky(0.1)
// ---------------------------------------------------------------------------
__global__ __launch_bounds__(256)
void conv_cin3_kernel(const float* __restrict__ x, const float* __restrict__ w,
                      const float* __restrict__ bias, _Float16* __restrict__ out,
                      int mode) {
  __shared__ float ws[64 * 27];
  __shared__ float bs[64];
  for (int i = threadIdx.x; i < 64 * 27; i += blockDim.x) ws[i] = w[i];
  for (int i = threadIdx.x; i < 64; i += blockDim.x) bs[i] = bias[i];
  __syncthreads();
  int p = blockIdx.x * blockDim.x + threadIdx.x;
  if (p >= NPOS) return;
  int n = p >> 16, y = (p >> 8) & 255, xc = p & 255;
  float xin[27];
#pragma unroll
  for (int tap = 0; tap < 9; ++tap) {
    int yy = y + tap / 3 - 1, xx = xc + tap % 3 - 1;
    bool inb = ((unsigned)yy < 256u) && ((unsigned)xx < 256u);
#pragma unroll
    for (int c = 0; c < 3; ++c)
      xin[c * 9 + tap] = inb ? x[((size_t)(n * 3 + c) << 16) + yy * WW + xx] : 0.0f;
  }
  _Float16* op = out + (size_t)pad_idx(n, y, xc) * 64;
  for (int co = 0; co < 64; ++co) {
    float a = bs[co];
#pragma unroll
    for (int k = 0; k < 27; ++k) a += ws[co * 27 + k] * xin[k];
    a = (mode == 1) ? (a > 0.0f ? a : 0.0f) : (a > 0.0f ? a : 0.1f * a);
    op[co] = (_Float16)a;
  }
}

// ---------------------------------------------------------------------------
// c2_2: 64 -> 3, padded f16 NHWC-64 in (halo zero => no bounds checks),
// fp32 [pos,3] out
// ---------------------------------------------------------------------------
__global__ __launch_bounds__(256)
void conv64to3_kernel(const _Float16* __restrict__ in, const float* __restrict__ w,
                      const float* __restrict__ bias, float* __restrict__ resid) {
  __shared__ float ws[3 * 64 * 9];
  for (int i = threadIdx.x; i < 3 * 64 * 9; i += blockDim.x) ws[i] = w[i];
  __syncthreads();
  int p = blockIdx.x * blockDim.x + threadIdx.x;
  if (p >= NPOS) return;
  int n = p >> 16, y = (p >> 8) & 255, xc = p & 255;
  float acc0 = bias[0], acc1 = bias[1], acc2 = bias[2];
#pragma unroll
  for (int tap = 0; tap < 9; ++tap) {
    int dy = tap / 3 - 1, dx = tap % 3 - 1;
    const _Float16* ip =
        in + (size_t)(n * PPLANE + (y + dy + 1) * PW + (xc + dx + 1)) * 64;
    for (int ci = 0; ci < 64; ++ci) {
      float xv = (float)ip[ci];
      acc0 += ws[(0 * 64 + ci) * 9 + tap] * xv;
      acc1 += ws[(1 * 64 + ci) * 9 + tap] * xv;
      acc2 += ws[(2 * 64 + ci) * 9 + tap] * xv;
    }
  }
  resid[(size_t)p * 3 + 0] = acc0;
  resid[(size_t)p * 3 + 1] = acc1;
  resid[(size_t)p * 3 + 2] = acc2;
}

// fea = h + fea1, written into feat channels 64..127 (free concat), all padded
__global__ __launch_bounds__(256)
void addcat_kernel(const _Float16* __restrict__ h, const _Float16* __restrict__ f1,
                   _Float16* __restrict__ feat) {
  size_t i = (size_t)blockIdx.x * blockDim.x + threadIdx.x;
  if (i >= (size_t)NPOS * 64) return;
  int p = (int)(i >> 6);
  int c = (int)(i & 63);
  int n = p >> 16, y = (p >> 8) & 255, xc = p & 255;
  size_t pp = (size_t)pad_idx(n, y, xc);
  feat[pp * 128 + 64 + c] = (_Float16)((float)h[pp * 64 + c] + (float)f1[pp * 64 + c]);
}

// KernelConv2D: out = sum_taps patches(x)*core + residual (core ch = c*25+ky*5+kx)
__global__ __launch_bounds__(256)
void kpn_final_kernel(const float* __restrict__ x, const _Float16* __restrict__ core,
                      const float* __restrict__ resid, float* __restrict__ outp) {
  int p = blockIdx.x * blockDim.x + threadIdx.x;
  if (p >= NPOS) return;
  int n = p >> 16, y = (p >> 8) & 255, xc = p & 255;
#pragma unroll
  for (int c = 0; c < 3; ++c) {
    float acc = resid[(size_t)p * 3 + c];
    const float* xp = x + ((size_t)(n * 3 + c) << 16);
#pragma unroll
    for (int t = 0; t < 25; ++t) {
      int yy = y + t / 5 - 2, xx = xc + t % 5 - 2;
      if (((unsigned)yy < 256u) && ((unsigned)xx < 256u))
        acc += xp[yy * WW + xx] * (float)core[(size_t)p * 80 + c * 25 + t];
    }
    outp[((size_t)(n * 3 + c) << 16) + y * WW + xc] = acc;
  }
}

// ---------------------------------------------------------------------------
extern "C" void kernel_launch(void* const* d_in, const int* in_sizes, int n_in,
                              void* d_out, int out_size, void* d_ws, size_t ws_size,
                              hipStream_t stream) {
  const float* x       = (const float*)d_in[0];
  const float* w_first = (const float*)d_in[1];
  const float* b_first = (const float*)d_in[2];
  const float* w_trunk = (const float*)d_in[3];
  const float* b_trunk = (const float*)d_in[4];
  const float* c1_w    = (const float*)d_in[5];
  const float* c1_b    = (const float*)d_in[6];
  const float* c2_w1   = (const float*)d_in[7];
  const float* c2_b1   = (const float*)d_in[8];
  const float* c2_w2   = (const float*)d_in[9];
  const float* c2_b2   = (const float*)d_in[10];
  const float* c3_w1   = (const float*)d_in[11];
  const float* c3_b1   = (const float*)d_in[12];
  const float* c3_w2   = (const float*)d_in[13];
  const float* c3_b2   = (const float*)d_in[14];
  const float* c3_w3   = (const float*)d_in[15];
  const float* c3_b3   = (const float*)d_in[16];
  float* outp = (float*)d_out;

  // ---- workspace carve (bump allocator, 256B aligned) ----
  char* ws = (char*)d_ws;
  size_t off = 0;
  auto alloc = [&](size_t bytes) -> char* {
    off = (off + 255) & ~(size_t)255;
    char* p = ws + off;
    off += bytes;
    return p;
  };
  const size_t PP = PADPOS;
  _Float16* h    = (_Float16*)alloc(PP * 64 * 2);   // trunk working (padded)
  _Float16* fea1 = (_Float16*)alloc(PP * 64 * 2);   // contiguous with h
  _Float16* t    = (_Float16*)alloc(PP * 64 * 2);
  _Float16* u    = (_Float16*)alloc(PP * 64 * 2);
  _Float16* feat = (_Float16*)alloc(PP * 128 * 2);  // [t-branch | fea] concat
  float*    resid = (float*)alloc((size_t)NPOS * 3 * 4);
  _Float16* core = (_Float16*)h;  // alias: h+fea1 dead by then; 262144*80*2 <= PP*128*2

  _Float16* wp_trunk = (_Float16*)alloc((size_t)32 * 36864 * 2);
  _Float16* wp_c32   = (_Float16*)alloc(36864 * 2);
  _Float16* wp_c33   = (_Float16*)alloc(36864 * 2);
  _Float16* wp_c21   = (_Float16*)alloc(73728 * 2);
  _Float16* wp_c1    = (_Float16*)alloc(92160 * 2);

  // ---- zero halos of all conv-input buffers (interior writes never touch halo)
  const int gridH = (PADPOS + 255) / 256;
  zero_halo_kernel<<<gridH, 256, 0, stream>>>(h, 64);
  zero_halo_kernel<<<gridH, 256, 0, stream>>>(fea1, 64);
  zero_halo_kernel<<<gridH, 256, 0, stream>>>(t, 64);
  zero_halo_kernel<<<gridH, 256, 0, stream>>>(u, 64);
  zero_halo_kernel<<<gridH, 256, 0, stream>>>(feat, 128);

  // ---- pack weights (tiny, every call; deterministic) ----
  for (int L = 0; L < 32; ++L)
    pack_w_kernel<<<144, 256, 0, stream>>>(w_trunk + (size_t)L * 36864,
                                           wp_trunk + (size_t)L * 36864, 64, 64, 4, 2);
  pack_w_kernel<<<144, 256, 0, stream>>>(c3_w2, wp_c32, 64, 64, 4, 2);
  pack_w_kernel<<<144, 256, 0, stream>>>(c3_w3, wp_c33, 64, 64, 4, 2);
  pack_w_kernel<<<288, 256, 0, stream>>>(c2_w1, wp_c21, 128, 64, 4, 4);
  pack_w_kernel<<<360, 256, 0, stream>>>(c1_w,  wp_c1,  128, 75, 5, 4);

  const int gridP = NPOS / 256;     // 1024 blocks, one thread per pixel
  const int gridW = 16384 / 8;      // 2048 blocks: 8 waves/block, 1 M-tile/wave (exact)

  // 1. fea1 = lrelu(conv_first(x))
  conv_cin3_kernel<<<gridP, 256, 0, stream>>>(x, w_first, b_first, fea1, 2);

  // 2. 16 residual blocks (in-place safe: conv input is t; res read == write pos)
  const _Float16* hin = fea1;
  for (int b = 0; b < 16; ++b) {
    conv3x3p_kernel<2, 4><<<gridW, 256, 0, stream>>>(
        hin, wp_trunk + (size_t)(2 * b) * 36864, b_trunk + (2 * b) * 64, 64,
        t, 64, 0, /*opad*/1, nullptr, 0, /*relu*/1);
    conv3x3p_kernel<2, 4><<<gridW, 256, 0, stream>>>(
        t, wp_trunk + (size_t)(2 * b + 1) * 36864, b_trunk + (2 * b + 1) * 64, 64,
        h, 64, 0, /*opad*/1, /*res*/hin, 64, /*relu*/0);
    hin = h;
  }

  // 3. feat[:,64:128] = h + fea1  (global residual + concat)
  addcat_kernel<<<(int)(((size_t)NPOS * 64 + 255) / 256), 256, 0, stream>>>(h, fea1, feat);

  // 4. t-branch: relu(conv3_1(x)) -> relu(conv3_2) -> conv3_3 into feat[:,0:64]
  conv_cin3_kernel<<<gridP, 256, 0, stream>>>(x, c3_w1, c3_b1, t, 1);
  conv3x3p_kernel<2, 4><<<gridW, 256, 0, stream>>>(
      t, wp_c32, c3_b2, 64, u, 64, 0, 1, nullptr, 0, 1);
  conv3x3p_kernel<2, 4><<<gridW, 256, 0, stream>>>(
      u, wp_c33, c3_b3, 64, feat, 128, 0, 1, nullptr, 0, 0);

  // 5. residual branch: relu(conv(feat,128->64)) then 64->3
  conv3x3p_kernel<4, 4><<<gridW, 256, 0, stream>>>(
      feat, wp_c21, c2_b1, 64, t, 64, 0, 1, nullptr, 0, 1);
  conv64to3_kernel<<<gridP, 256, 0, stream>>>(t, c2_w2, c2_b2, resid);

  // 6. core = conv(feat, 128->75 padded to 80), flat layout (not a conv input)
  conv3x3p_kernel<4, 5><<<gridW, 256, 0, stream>>>(
      feat, wp_c1, c1_b, 75, core, 80, 0, /*opad*/0, nullptr, 0, 0);

  // 7. KPN tap-sum + residual -> NCHW fp32 output
  kpn_final_kernel<<<gridP, 256, 0, stream>>>(x, core, resid, outp);
}
